// HyperGraphAttentionNetwork_23862838297173
// MI455X (gfx1250) — matrix operand order repaired and checked by reading, not compiled, not run on hardware
//
#include <hip/hip_runtime.h>
#include <hip/hip_bf16.h>

typedef __attribute__((ext_vector_type(16))) __bf16 v16bf;
typedef __attribute__((ext_vector_type(8)))  __bf16 v8bf;
typedef __attribute__((ext_vector_type(8)))  float  v8f;
typedef __attribute__((ext_vector_type(4)))  float  v4f;

#define GAT_H 8
#define GAT_O 32
#define GAT_HO 256
#define NEG_SLOPE 0.2f

// ---------------- utility kernels ----------------

__global__ void fill_f32(float* __restrict__ p, float v, long n) {
    long i = (long)blockIdx.x * blockDim.x + threadIdx.x;
    if (i < n) p[i] = v;
}

__global__ void init_u32(unsigned* p, unsigned v) {
    if (blockIdx.x == 0 && threadIdx.x == 0) *p = v;
}

// order-preserving float <-> uint encoding for atomicMax on floats
__device__ __forceinline__ unsigned f2ord(float f) {
    unsigned u = __float_as_uint(f);
    return (u & 0x80000000u) ? ~u : (u | 0x80000000u);
}
__device__ __forceinline__ float ord2f(unsigned u) {
    u = (u & 0x80000000u) ? (u & 0x7fffffffu) : ~u;
    return __uint_as_float(u);
}

// ---------------- WMMA GEMM: h_prime[N, H*O] = act(A[N,K]) x W[H,K,O] ----------------
// Block: 256 threads = 8 wave32 -> 32(M) x 128(N) tile. Each wave computes two
// 16x16 tiles sharing one A fragment (2x v_wmma_f32_16x16x32_bf16 per K-step).
// A tile: async-staged raw f32 (global_load_async_to_lds_b128, ASYNCcnt),
// converted to bf16 (+optional ELU) at fragment read via v_cvt_pk_bf16_f32.
// B tile: staged bf16 TRANSPOSED [n][k] so a lane's fragment is one contiguous
// 16-element k-run -> two ds_load_b128 per fragment.

__global__ __launch_bounds__(256)
void gemm_hprime_wmma(const float* __restrict__ A, const float* __restrict__ W,
                      float* __restrict__ C, int N, int K, int applyElu)
{
    __shared__ __align__(16) float  Asf[32][36];   // raw f32 A tile (stride 144B, 16B-aligned rows)
    __shared__ __align__(16) __bf16 Bs[128][48];   // bf16 B tile transposed (stride 96B, 16B-aligned)

    const int tid  = threadIdx.x;
    const int wave = tid >> 5;
    const int lane = tid & 31;
    const int r    = lane & 15;
    const int hsel = lane >> 4;

    const int rowBase = blockIdx.x * 32;     // M tile origin
    const int colBase = blockIdx.y * 128;    // N tile origin within H*O = 256
    const int mBase   = (wave >> 2) * 16;    // wave's M subtile (0 or 16)
    const int nBase   = (wave & 3) * 32;     // wave's N subtile pair origin

    const unsigned asfBase = (unsigned)(size_t)&Asf[0][0];

    v8f acc0 = {}, acc1 = {};

#pragma unroll 2
    for (int k0 = 0; k0 < K; k0 += 32) {
        // ---- async-stage A tile (32x32 f32): 1 b128 per thread, 1 ASYNCcnt/wave ----
        {
            int m  = tid >> 3;                 // 0..31
            int kq = (tid & 7) * 4;            // 0,4,...,28
            int gr = rowBase + m;
            if (gr >= N) gr = N - 1;           // clamp (avoids OOB; rows beyond N unused)
            const float* gp = A + (long)gr * K + (k0 + kq);
            unsigned ldsOff = asfBase + (unsigned)(m * 36 + kq) * 4u;
            asm volatile("global_load_async_to_lds_b128 %0, %1, off"
                         :: "v"(ldsOff), "v"(gp) : "memory");
        }
        // ---- sync-stage B tile (32k x 128n) as bf16, transposed [n][k] ----
        for (int t = tid; t < 32 * 128; t += 256) {
            int kk = t >> 7, jj = t & 127;     // consecutive lanes -> consecutive o (coalesced)
            int j = colBase + jj;
            int h = j >> 5, o = j & 31;
            Bs[jj][kk] = (__bf16)W[((long)h * K + (k0 + kk)) * GAT_O + o];
        }
        asm volatile("s_wait_asynccnt 0x0" ::: "memory");
        __syncthreads();

        // ---- A fragment: one contiguous 16-f32 run -> 4x b128 LDS loads + cvt ----
        const float* asRow = &Asf[mBase + r][0];
        v4f aw0 = *(const v4f*)(asRow + hsel * 16);
        v4f aw1 = *(const v4f*)(asRow + hsel * 16 + 4);
        v4f aw2 = *(const v4f*)(asRow + hsel * 16 + 8);
        v4f aw3 = *(const v4f*)(asRow + hsel * 16 + 12);
        v16bf af;
#pragma unroll
        for (int j = 0; j < 16; ++j) {
            float x = (j < 4) ? aw0[j] : (j < 8) ? aw1[j & 3] : (j < 12) ? aw2[j & 3] : aw3[j & 3];
            if (applyElu) x = (x > 0.f) ? x : (__expf(x) - 1.f);
            af[j] = (__bf16)x;
        }

        // ---- B fragments: one contiguous 16-bf16 k-run each -> 2x b128 per tile ----
        v8bf b0lo = *(const v8bf*)&Bs[nBase + r][hsel * 16];
        v8bf b0hi = *(const v8bf*)&Bs[nBase + r][hsel * 16 + 8];
        v8bf b1lo = *(const v8bf*)&Bs[nBase + 16 + r][hsel * 16];
        v8bf b1hi = *(const v8bf*)&Bs[nBase + 16 + r][hsel * 16 + 8];
        v16bf bf0 = __builtin_shufflevector(b0lo, b0hi, 0, 1, 2, 3, 4, 5, 6, 7, 8, 9, 10, 11, 12, 13, 14, 15);
        v16bf bf1 = __builtin_shufflevector(b1lo, b1hi, 0, 1, 2, 3, 4, 5, 6, 7, 8, 9, 10, 11, 12, 13, 14, 15);

        acc0 = __builtin_amdgcn_wmma_f32_16x16x32_bf16(false, af, false, bf0, (short)0, acc0, false, false);
        acc1 = __builtin_amdgcn_wmma_f32_16x16x32_bf16(false, af, false, bf1, (short)0, acc1, false, false);
        __syncthreads();
    }

    // store D: VGPR v -> row (v + 8*hsel), col (lane&15)
#pragma unroll
    for (int v = 0; v < 8; ++v) {
        int row = rowBase + mBase + hsel * 8 + v;
        if (row < N) {
            C[(long)row * GAT_HO + colBase + nBase + r]      = acc0[v];
            C[(long)row * GAT_HO + colBase + nBase + 16 + r] = acc1[v];
        }
    }
}

// ---------------- attention scores: s[n,h] = <h_prime[n,h,:], a[h,:]> ----------------

__global__ void attn_scores(const float* __restrict__ hp, const float* __restrict__ aS,
                            const float* __restrict__ aT, float* __restrict__ sS,
                            float* __restrict__ sT, int N)
{
    int i = blockIdx.x * blockDim.x + threadIdx.x;   // i = n*H + h
    if (i >= N * GAT_H) return;
    int h = i & (GAT_H - 1);
    const float* row = hp + (long)i * GAT_O;
    float s = 0.f, t = 0.f;
#pragma unroll
    for (int o = 0; o < GAT_O; ++o) {
        float v = row[o];
        s += v * aS[h * GAT_O + o];
        t += v * aT[h * GAT_O + o];
    }
    sS[i] = s;
    sT[i] = t;
}

// ---------------- global max of leaky_relu(e) over all edges/heads ----------------

__global__ void edge_max(const float* __restrict__ sS, const float* __restrict__ sT,
                         const int* __restrict__ src, const int* __restrict__ trg,
                         unsigned* gmax, int E)
{
    int e = blockIdx.x * blockDim.x + threadIdx.x;
    float m = -__builtin_inff();
    if (e < E) {
        int s = src[e], t = trg[e];
#pragma unroll
        for (int h = 0; h < GAT_H; ++h) {
            float v = sS[s * GAT_H + h] + sT[t * GAT_H + h];
            v = (v > 0.f) ? v : v * NEG_SLOPE;
            m = fmaxf(m, v);
        }
    }
    __shared__ float red[256];
    red[threadIdx.x] = m;
    __syncthreads();
    for (int off = 128; off > 0; off >>= 1) {
        if (threadIdx.x < off) red[threadIdx.x] = fmaxf(red[threadIdx.x], red[threadIdx.x + off]);
        __syncthreads();
    }
    if (threadIdx.x == 0) atomicMax(gmax, f2ord(red[0]));
}

// ---------------- exp(e - max) and per-target denominator ----------------

__global__ void edge_exp(const float* __restrict__ sS, const float* __restrict__ sT,
                         const int* __restrict__ src, const int* __restrict__ trg,
                         const unsigned* __restrict__ gmax, float* __restrict__ expE,
                         float* __restrict__ denom, int E)
{
    int e = blockIdx.x * blockDim.x + threadIdx.x;
    if (e >= E) return;
    float mx = ord2f(*gmax);
    int s = src[e], t = trg[e];
#pragma unroll
    for (int h = 0; h < GAT_H; ++h) {
        float v = sS[s * GAT_H + h] + sT[t * GAT_H + h];
        v = (v > 0.f) ? v : v * NEG_SLOPE;
        float ex = __expf(v - mx);
        expE[(long)e * GAT_H + h] = ex;
        atomicAdd(&denom[t * GAT_H + h], ex);
    }
}

// ---------------- streaming edge aggregation (memory-bound stage) ----------------
// thread t covers (edge e = t>>8, feature rem = t&255); consecutive threads read
// one contiguous 1KB h_prime row and scatter with global_atomic_add_f32.

__global__ void edge_aggregate(const float* __restrict__ hp, const float* __restrict__ expE,
                               const float* __restrict__ denom, const int* __restrict__ src,
                               const int* __restrict__ trg, float* __restrict__ out, long total)
{
    long t = (long)blockIdx.x * blockDim.x + threadIdx.x;
    if (t >= total) return;
    long e = t >> 8;
    int rem = (int)(t & 255);
    int h = rem >> 5;
    int s = src[e], tg = trg[e];
    float alpha = expE[e * GAT_H + h] / (denom[tg * GAT_H + h] + 1e-16f);
    atomicAdd(&out[(long)tg * GAT_HO + rem], hp[(long)s * GAT_HO + rem] * alpha);
}

// ---------------- mean over heads (final layer) ----------------

__global__ void mean_heads(const float* __restrict__ acc, float* __restrict__ emb, int nUser)
{
    int i = blockIdx.x * blockDim.x + threadIdx.x;   // i = n*32 + o
    if (i >= nUser * GAT_O) return;
    int n = i >> 5, o = i & 31;
    float s = 0.f;
#pragma unroll
    for (int h = 0; h < GAT_H; ++h) s += acc[(long)n * GAT_HO + h * GAT_O + o];
    emb[i] = s * 0.125f;
}

// ---------------- concat + FC + log_softmax ----------------

__global__ void fc_logsoftmax(const float* __restrict__ emb0, const float* __restrict__ emb1,
                              const float* __restrict__ fcw, const float* __restrict__ fcb,
                              float* __restrict__ out, int nUser)
{
    int n = blockIdx.x * blockDim.x + threadIdx.x;
    if (n >= nUser) return;
    float l0 = fcb[0], l1 = fcb[1];
#pragma unroll
    for (int i = 0; i < 32; ++i) {
        float v = emb0[n * 32 + i];
        l0 += v * fcw[i * 2 + 0];
        l1 += v * fcw[i * 2 + 1];
    }
#pragma unroll
    for (int i = 0; i < 32; ++i) {
        float v = emb1[n * 32 + i];
        l0 += v * fcw[(32 + i) * 2 + 0];
        l1 += v * fcw[(32 + i) * 2 + 1];
    }
    float m = fmaxf(l0, l1);
    float lse = m + __logf(__expf(l0 - m) + __expf(l1 - m));
    out[n * 2 + 0] = l0 - lse;
    out[n * 2 + 1] = l1 - lse;
}

// ---------------- host orchestration ----------------

extern "C" void kernel_launch(void* const* d_in, const int* in_sizes, int n_in,
                              void* d_out, int out_size, void* d_ws, size_t ws_size,
                              hipStream_t stream) {
    const int F0 = 128;
    const int N = in_sizes[0] / F0;       // 60000
    const int E = in_sizes[2];            // 800000
    const int nUser = out_size / 2;       // 50000

    // workspace carve-out
    char* ws = (char*)d_ws;
    size_t off = 0;
    auto carve = [&](size_t bytes) -> void* {
        void* p = ws + off;
        off += (bytes + 255) & ~(size_t)255;
        return p;
    };
    float*    bufA  = (float*)carve((size_t)N * GAT_HO * 4);  // h_prime
    float*    bufB  = (float*)carve((size_t)N * GAT_HO * 4);  // layer accumulator / next input
    float*    expE  = (float*)carve((size_t)E * GAT_H * 4);
    float*    attnS = (float*)carve((size_t)N * GAT_H * 4);
    float*    attnT = (float*)carve((size_t)N * GAT_H * 4);
    float*    denom = (float*)carve((size_t)N * GAT_H * 4);
    float*    emb0  = (float*)carve((size_t)nUser * GAT_O * 4);
    float*    emb1  = (float*)carve((size_t)nUser * GAT_O * 4);
    unsigned* gmax  = (unsigned*)carve(256);
    (void)ws_size; (void)n_in;

    for (int g = 0; g < 2; ++g) {
        const float* h   = (const float*)d_in[g];
        const int*   src = (const int*)d_in[2 + 2 * g];
        const int*   trg = (const int*)d_in[3 + 2 * g];
        float*       emb = (g == 0) ? emb0 : emb1;

        for (int l = 0; l < 2; ++l) {
            const float* w  = (const float*)d_in[6 + 6 * g + 3 * l];
            const float* aS = (const float*)d_in[7 + 6 * g + 3 * l];
            const float* aT = (const float*)d_in[8 + 6 * g + 3 * l];
            const int K = (l == 0) ? F0 : GAT_HO;
            const float* Ain = (l == 0) ? h : bufB;   // layer-2 reads layer-1 acc (ELU in GEMM)

            dim3 ggrid((N + 31) / 32, GAT_HO / 128);
            gemm_hprime_wmma<<<ggrid, 256, 0, stream>>>(Ain, w, bufA, N, K, l == 0 ? 0 : 1);

            attn_scores<<<(N * GAT_H + 255) / 256, 256, 0, stream>>>(bufA, aS, aT, attnS, attnT, N);

            // reset accumulator (safe: GEMM already consumed bufB), denom, global max
            fill_f32<<<(int)(((long)N * GAT_HO + 255) / 256), 256, 0, stream>>>(bufB, 0.f, (long)N * GAT_HO);
            fill_f32<<<(N * GAT_H + 255) / 256, 256, 0, stream>>>(denom, 0.f, (long)N * GAT_H);
            init_u32<<<1, 32, 0, stream>>>(gmax, 0u);

            edge_max<<<(E + 255) / 256, 256, 0, stream>>>(attnS, attnT, src, trg, gmax, E);
            edge_exp<<<(E + 255) / 256, 256, 0, stream>>>(attnS, attnT, src, trg, gmax, expE, denom, E);

            long total = (long)E * GAT_HO;
            edge_aggregate<<<(int)((total + 255) / 256), 256, 0, stream>>>(
                bufA, expE, denom, src, trg, bufB, total);
        }

        mean_heads<<<(nUser * GAT_O + 255) / 256, 256, 0, stream>>>(bufB, emb, nUser);
    }

    fc_logsoftmax<<<(nUser + 255) / 256, 256, 0, stream>>>(
        emb0, emb1, (const float*)d_in[18], (const float*)d_in[19], (float*)d_out, nUser);
}
